// MSR_70171175682719
// MI455X (gfx1250) — compile-verified, matrix operand order checked by reading, not run on hardware
//
#include <hip/hip_runtime.h>
#include <hip/hip_bf16.h>
#include <math.h>
#include <stdint.h>

// ---------------------------------------------------------------------------
// Types for CDNA5 WMMA (wave32): bf16 16x16x32 -> f32 accumulators
// ---------------------------------------------------------------------------
typedef __bf16 bf16_t;
typedef __attribute__((ext_vector_type(16))) __bf16 v16bf;
typedef __attribute__((ext_vector_type(8)))  __bf16 v8bf;
typedef __attribute__((ext_vector_type(8)))  float  v8f;
typedef __attribute__((ext_vector_type(4)))  unsigned int v4u;
typedef __attribute__((ext_vector_type(8)))  int v8i;
typedef __attribute__((ext_vector_type(4)))  int v4i;

#define WMMA_BF16(a, b, c) \
  __builtin_amdgcn_wmma_f32_16x16x32_bf16(false, (a), false, (b), (short)0, (c), false, false)

#if defined(__HIP_DEVICE_COMPILE__) && __has_builtin(__builtin_amdgcn_tensor_load_to_lds)
#define HAVE_TDM 1
#else
#define HAVE_TDM 0
#endif

// Problem constants (from the reference)
constexpr int Bc  = 2;
constexpr int Sc  = 2048;
constexpr int Dc  = 1024;
constexpr int Hc  = 8;
constexpr int DHc = 128;   // D / H

// ---------------------------------------------------------------------------
// Fragment loader. Per the CDNA5 ISA 16-bit A-matrix 16x32 layout:
//   lane L -> row M = L%16
//   halfword e in [0,8)  -> K = k0 + (L>=16 ? 8:0) + e         (contiguous 8)
//   halfword e in [8,16) -> K = k0 + 16 + (L>=16 ? 8:0) + e-8  (contiguous 8)
// B-matrix (32x16) mirrors this with lane -> column N. So for row-major
// operands with K contiguous, each lane issues two 16-byte loads.
// ---------------------------------------------------------------------------
__device__ __forceinline__ v16bf load_frag(const bf16_t* base, int row, int ld,
                                           int k0, int lane) {
  const bf16_t* p = base + (size_t)row * ld + k0 + ((lane & 16) ? 8 : 0);
  union { v16bf v; v8bf h[2]; } u;
  u.h[0] = *(const v8bf*)(p);
  u.h[1] = *(const v8bf*)(p + 16);
  return u.v;
}

// ---------------------------------------------------------------------------
// Converters
// ---------------------------------------------------------------------------
__global__ void cvt_f32_bf16(const float* __restrict__ in, bf16_t* __restrict__ out, int n) {
  for (int i = blockIdx.x * blockDim.x + threadIdx.x; i < n; i += gridDim.x * blockDim.x)
    out[i] = (bf16_t)in[i];
}

// Wt[n*D + k] = (bf16) W[k*D + n]  -- B operand becomes row-major-in-K
__global__ void transpose_cvt(const float* __restrict__ W, bf16_t* __restrict__ Wt, int D) {
  int n = blockIdx.x * 16 + threadIdx.x;
  int k = blockIdx.y * 16 + threadIdx.y;
  if (n < D && k < D) Wt[(size_t)n * D + k] = (bf16_t)W[(size_t)k * D + n];
}

// ---------------------------------------------------------------------------
// GEMM epilogue: store one 16x16 f32 C tile per documented C layout:
//   VGPR vr: lanes 0-15 -> (M=vr, N=lane), lanes 16-31 -> (M=vr+8, N=lane-16)
// ---------------------------------------------------------------------------
__device__ __forceinline__ void store_tile(const v8f& c, int r0, int c0,
                                           float* outF, bf16_t* outB,
                                           int N, float scale, int mode, int lane) {
  const int mloc = (lane & 16) ? 8 : 0;
  const int col  = c0 + (lane & 15);
#pragma unroll
  for (int vr = 0; vr < 8; ++vr) {
    const int row = r0 + vr + mloc;
    float v = c[vr] * scale;
    if (mode == 0) {                       // plain f32 row-major (final output)
      outF[(size_t)row * N + col] = v;
    } else if (mode == 1) {                // bf16 [B,H,S,DH]  (Q scaled / K)
      int b = row >> 11, s = row & (Sc - 1);
      int h = col >> 7,  dh = col & (DHc - 1);
      outB[((((size_t)b * Hc + h) * Sc) + s) * DHc + dh] = (bf16_t)v;
    } else if (mode == 2) {                // bf16 V^T [B,H,DH,S]
      int b = row >> 11, s = row & (Sc - 1);
      int h = col >> 7,  dh = col & (DHc - 1);
      outB[(((size_t)b * Hc + h) * DHc + dh) * Sc + s] = (bf16_t)v;
    } else {                               // f32 silu(x) [B,S,D]
      float sv = v / (1.0f + __expf(-v));
      outF[(size_t)row * N + col] = sv;
    }
  }
}

#if HAVE_TDM
// ---------------------------------------------------------------------------
// Tensor Data Mover: issue a 2-D tile load Global->LDS.
// D# packing per CDNA5 ISA:
//   group0: [1:0]=count=1, [63:32]=lds_addr, [120:64]=global_addr, [127:126]=2
//   group1: data_size=1(2B); pad_interval=4 (32 dwords); pad_amount=3 (4 dwords)
//           => LDS row pitch = 36 dwords = 72 bf16 elements per 64-elem row.
//   tensor_dim0/dim0_stride = K row length; tile = tile_d0 x tile_d1.
// ---------------------------------------------------------------------------
__device__ __forceinline__ void tdm_load_2d(uint32_t lds_addr, const bf16_t* gptr,
                                            uint32_t tensor_d0, uint32_t tensor_d1,
                                            uint32_t tile_d0, uint32_t tile_d1,
                                            uint64_t stride_elems) {
  uint64_t ga = (uint64_t)(uintptr_t)gptr;
  v4u g0;
  g0[0] = 1u;                                            // count=1
  g0[1] = lds_addr;                                      // LDS byte address
  g0[2] = (uint32_t)(ga & 0xFFFFFFFFu);                  // global_addr[31:0]
  g0[3] = (uint32_t)((ga >> 32) & 0x01FFFFFFu) | (2u << 30);  // addr[56:32], type=2
  v8i g1;
  g1[0] = (int)((1u << 16) | (1u << 20) | (4u << 22) | (3u << 25));
  //            data_size=2B | pad_enable | pad_interval=32dw | pad_amount=4dw
  g1[1] = (int)((tensor_d0 & 0xFFFFu) << 16);
  g1[2] = (int)(((tensor_d0 >> 16) & 0xFFFFu) | ((tensor_d1 & 0xFFFFu) << 16));
  g1[3] = (int)(((tensor_d1 >> 16) & 0xFFFFu) | ((tile_d0 & 0xFFFFu) << 16));
  g1[4] = (int)(tile_d1 & 0xFFFFu);                      // tile_dim2 = 0
  g1[5] = (int)(uint32_t)(stride_elems & 0xFFFFFFFFu);
  g1[6] = (int)(uint32_t)((stride_elems >> 32) & 0xFFFFu);
  g1[7] = 0;
  v4i z4 = {0, 0, 0, 0};
#if defined(__clang_major__) && __clang_major__ >= 23
  v8i z8 = {0, 0, 0, 0, 0, 0, 0, 0};
  __builtin_amdgcn_tensor_load_to_lds(g0, g1, z4, z4, z8, 0);
#else
  __builtin_amdgcn_tensor_load_to_lds(g0, g1, z4, z4, 0);
#endif
}

// ---------------------------------------------------------------------------
// GEMM with TDM double-buffered LDS staging.
// Block: 8 waves (2 row-groups x 4 col-groups); block tile 64(M) x 128(N).
// K step 64: A slab 64x64, B slab 128x64 -> LDS (pitch 72), wave tile 32x32.
// Wave 0 drives the TDM pipeline; TENSORcnt + barriers synchronize.
// ---------------------------------------------------------------------------
__global__ __launch_bounds__(256) void gemm_bf16_k(
    const bf16_t* __restrict__ A, const bf16_t* __restrict__ Bt,
    float* __restrict__ outF, bf16_t* __restrict__ outB,
    int M, int N, int Kd, float scale, int mode) {
  constexpr int MB = 64, NB = 128, KB = 64, PITCH = 72;
  __shared__ __align__(16) bf16_t Ab[2][MB * PITCH];
  __shared__ __align__(16) bf16_t Bb[2][NB * PITCH];

  const int lane = threadIdx.x & 31;
  const int wave = threadIdx.x >> 5;
  const int wr = wave >> 2, wc = wave & 3;
  const int mBase = blockIdx.y * MB;
  const int nBase = blockIdx.x * NB;
  const int m0 = wr * 32;                 // local row of this wave's 32x32 tile
  const int n0 = wc * 32;                 // local col

  const int niter = Kd / KB;
  if (wave == 0) {   // prologue: stage k-slab 0 into buffer 0
    tdm_load_2d((uint32_t)(uintptr_t)&Ab[0][0], A + (size_t)mBase * Kd,
                (uint32_t)Kd, (uint32_t)(M - mBase), KB, MB, (uint64_t)Kd);
    tdm_load_2d((uint32_t)(uintptr_t)&Bb[0][0], Bt + (size_t)nBase * Kd,
                (uint32_t)Kd, (uint32_t)(N - nBase), KB, NB, (uint64_t)Kd);
  }

  v8f c00 = {}, c01 = {}, c10 = {}, c11 = {};
  const int rA = m0 + (lane & 15);
  const int rB = n0 + (lane & 15);

  for (int it = 0; it < niter; ++it) {
    if (wave == 0) {
      if (it + 1 < niter) {   // kick next slab into the other buffer
        const int kn = (it + 1) * KB;
        const int nb = (it + 1) & 1;
        tdm_load_2d((uint32_t)(uintptr_t)&Ab[nb][0], A + (size_t)mBase * Kd + kn,
                    (uint32_t)(Kd - kn), (uint32_t)(M - mBase), KB, MB, (uint64_t)Kd);
        tdm_load_2d((uint32_t)(uintptr_t)&Bb[nb][0], Bt + (size_t)nBase * Kd + kn,
                    (uint32_t)(Kd - kn), (uint32_t)(N - nBase), KB, NB, (uint64_t)Kd);
        __builtin_amdgcn_s_wait_tensorcnt(2);  // current slab's 2 loads done
      } else {
        __builtin_amdgcn_s_wait_tensorcnt(0);
      }
    }
    __syncthreads();
    const bf16_t* As = &Ab[it & 1][0];
    const bf16_t* Bs = &Bb[it & 1][0];
#pragma unroll
    for (int kk = 0; kk < KB; kk += 32) {
      v16bf a0 = load_frag(As, rA,      PITCH, kk, lane);
      v16bf a1 = load_frag(As, rA + 16, PITCH, kk, lane);
      v16bf b0 = load_frag(Bs, rB,      PITCH, kk, lane);
      v16bf b1 = load_frag(Bs, rB + 16, PITCH, kk, lane);
      c00 = WMMA_BF16(a0, b0, c00);
      c01 = WMMA_BF16(a0, b1, c01);
      c10 = WMMA_BF16(a1, b0, c10);
      c11 = WMMA_BF16(a1, b1, c11);
    }
    __syncthreads();   // LDS reads done before TDM overwrites this buffer
  }

  store_tile(c00, mBase + m0,      nBase + n0,      outF, outB, N, scale, mode, lane);
  store_tile(c01, mBase + m0,      nBase + n0 + 16, outF, outB, N, scale, mode, lane);
  store_tile(c10, mBase + m0 + 16, nBase + n0,      outF, outB, N, scale, mode, lane);
  store_tile(c11, mBase + m0 + 16, nBase + n0 + 16, outF, outB, N, scale, mode, lane);
}

#else  // !HAVE_TDM: direct-from-global fallback (previous round's kernel)

__global__ __launch_bounds__(256) void gemm_bf16_k(
    const bf16_t* __restrict__ A, const bf16_t* __restrict__ Bt,
    float* __restrict__ outF, bf16_t* __restrict__ outB,
    int M, int N, int Kd, float scale, int mode) {
  const int lane = threadIdx.x & 31;
  const int wave = threadIdx.x >> 5;
  const int wr = wave >> 2, wc = wave & 3;
  const int m0 = blockIdx.y * 64 + wr * 32;
  const int n0 = blockIdx.x * 128 + wc * 32;
  if (n0 >= N || m0 >= M) return;

  v8f c00 = {}, c01 = {}, c10 = {}, c11 = {};
  const int rA0 = m0 + (lane & 15), rA1 = rA0 + 16;
  const int rB0 = n0 + (lane & 15), rB1 = rB0 + 16;

  for (int k0 = 0; k0 < Kd; k0 += 32) {
    if (k0 + 64 <= Kd) {
      __builtin_prefetch(A  + (size_t)rA0 * Kd + k0 + 64, 0, 1);
      __builtin_prefetch(Bt + (size_t)rB0 * Kd + k0 + 64, 0, 1);
    }
    v16bf a0 = load_frag(A,  rA0, Kd, k0, lane);
    v16bf a1 = load_frag(A,  rA1, Kd, k0, lane);
    v16bf b0 = load_frag(Bt, rB0, Kd, k0, lane);
    v16bf b1 = load_frag(Bt, rB1, Kd, k0, lane);
    c00 = WMMA_BF16(a0, b0, c00);
    c01 = WMMA_BF16(a0, b1, c01);
    c10 = WMMA_BF16(a1, b0, c10);
    c11 = WMMA_BF16(a1, b1, c11);
  }
  store_tile(c00, m0,      n0,      outF, outB, N, scale, mode, lane);
  store_tile(c01, m0,      n0 + 16, outF, outB, N, scale, mode, lane);
  store_tile(c10, m0 + 16, n0,      outF, outB, N, scale, mode, lane);
  store_tile(c11, m0 + 16, n0 + 16, outF, outB, N, scale, mode, lane);
}
#endif

// ---------------------------------------------------------------------------
// Retention: per wave, one (b,h) x 16-row tile. Causal loop over 32-key
// blocks: 8 WMMAs for Q*K^T, decay+mask in f32, LDS round-trip to convert the
// score tile from C layout to A layout, 8 WMMAs for P*V (V pre-transposed).
// Epilogue: GroupNorm over DH via shfl_xor reductions, * silu(G), bf16 store.
// ---------------------------------------------------------------------------
__global__ __launch_bounds__(128) void retention_k(
    const bf16_t* __restrict__ Qb, const bf16_t* __restrict__ Kb,
    const bf16_t* __restrict__ Vtb, const float* __restrict__ Gs,
    bf16_t* __restrict__ Yg) {
  const int lane = threadIdx.x & 31;
  const int wv   = threadIdx.x >> 5;           // 4 waves / block
  const int gw   = blockIdx.x * 4 + wv;
  const int ntiles = Sc / 16;
  const int n_t = gw % ntiles;
  const int bh  = gw / ntiles;                 // 0 .. B*H-1
  const int h   = bh & (Hc - 1);
  const int b   = bh >> 3;
  const int n0  = n_t * 16;

  const bf16_t* q  = Qb  + (size_t)bh * Sc * DHc;
  const bf16_t* k  = Kb  + (size_t)bh * Sc * DHc;
  const bf16_t* vt = Vtb + (size_t)bh * DHc * Sc;

  __shared__ __align__(16) bf16_t Pbuf[4][16 * 32];
  bf16_t* P = Pbuf[wv];

  // Q fragments for the whole 16x128 row tile (Q already scaled by 1/sqrt(DH))
  v16bf aQ[4];
  const int rq = n0 + (lane & 15);
#pragma unroll
  for (int i = 0; i < 4; ++i) aQ[i] = load_frag(q, rq, DHc, 32 * i, lane);

  v8f acc[8];
#pragma unroll
  for (int j = 0; j < 8; ++j) acc[j] = v8f{};

  const float log2g = log2f(1.0f - exp2f(-5.0f - (float)h));
  const int mloc = (lane & 16) ? 8 : 0;
  const int ncol = lane & 15;

  for (int m0 = 0; m0 <= n0 + 15; m0 += 32) {
    // ---- scores S0 (keys m0..m0+15), S1 (keys m0+16..m0+31) ----
    v8f s0 = {}, s1 = {};
    const int rk0 = m0 + (lane & 15);
#pragma unroll
    for (int i = 0; i < 4; ++i) {
      v16bf bk = load_frag(k, rk0, DHc, 32 * i, lane);
      s0 = WMMA_BF16(aQ[i], bk, s0);
    }
    const bool doS1 = (m0 < n0);               // wave-uniform
    if (doS1) {
      const int rk1 = rk0 + 16;
#pragma unroll
      for (int i = 0; i < 4; ++i) {
        v16bf bk = load_frag(k, rk1, DHc, 32 * i, lane);
        s1 = WMMA_BF16(aQ[i], bk, s1);
      }
    }

    // ---- decay * causal mask, pack bf16 P tile (16x32) into LDS ----
#pragma unroll
    for (int vr = 0; vr < 8; ++vr) {
      const int n = n0 + vr + mloc;
      float d0 = (float)(n - (m0 + ncol));
      float v0 = (d0 >= 0.0f) ? s0[vr] * exp2f(log2g * d0) : 0.0f;
      P[(vr + mloc) * 32 + ncol] = (bf16_t)v0;
      float d1 = (float)(n - (m0 + 16 + ncol));
      float v1 = (doS1 && d1 >= 0.0f) ? s1[vr] * exp2f(log2g * d1) : 0.0f;
      P[(vr + mloc) * 32 + 16 + ncol] = (bf16_t)v1;
    }

    // ---- P (A layout) x V^T fragments: ret[16,128] += P[16,32] @ V[32,128]
    v16bf aP = load_frag(P, (lane & 15), 32, 0, lane);
#pragma unroll
    for (int j = 0; j < 8; ++j) {
      v16bf bV = load_frag(vt, 16 * j + (lane & 15), Sc, m0, lane);
      acc[j] = WMMA_BF16(aP, bV, acc[j]);
    }
  }

  // ---- GroupNorm over DH per row, then * silu(G), store bf16 Yg[B,S,D] ----
  const float inv_dh = 1.0f / (float)DHc;
#pragma unroll
  for (int vr = 0; vr < 8; ++vr) {
    const int n = n0 + vr + mloc;
    float sum = 0.0f, ssq = 0.0f;
#pragma unroll
    for (int j = 0; j < 8; ++j) { float x = acc[j][vr]; sum += x; ssq += x * x; }
    // reduce across the 16 lanes holding this row (xor of bits 0..3)
    for (int msk = 1; msk < 16; msk <<= 1) {
      sum += __shfl_xor(sum, msk, 32);
      ssq += __shfl_xor(ssq, msk, 32);
    }
    const float mean = sum * inv_dh;
    const float var  = ssq * inv_dh - mean * mean;
    const float rstd = rsqrtf(var + 1e-6f);
    const size_t rowoff = ((size_t)b * Sc + n) * Dc + h * DHc;
#pragma unroll
    for (int j = 0; j < 8; ++j) {
      float y = (acc[j][vr] - mean) * rstd;
      size_t idx = rowoff + 16 * j + ncol;
      Yg[idx] = (bf16_t)(y * Gs[idx]);
    }
  }
}

// ---------------------------------------------------------------------------
// Host-side orchestration (graph-capture safe: kernels only, all on stream)
// ---------------------------------------------------------------------------
extern "C" void kernel_launch(void* const* d_in, const int* in_sizes, int n_in,
                              void* d_out, int out_size, void* d_ws, size_t ws_size,
                              hipStream_t stream) {
  const float* X  = (const float*)d_in[0];
  const float* Wq = (const float*)d_in[1];
  const float* Wk = (const float*)d_in[2];
  const float* Wv = (const float*)d_in[3];
  const float* Wg = (const float*)d_in[4];
  const float* Wo = (const float*)d_in[5];
  float* Out = (float*)d_out;

  const int M = Bc * Sc;                 // 4096 token rows
  const size_t nX = (size_t)M * Dc;      // 4,194,304
  const size_t nW = (size_t)Dc * Dc;     // 1,048,576

  // workspace layout (bytes)
  char* ws = (char*)d_ws;
  size_t off = 0;
  bf16_t* Xb  = (bf16_t*)(ws + off); off += nX * 2;
  bf16_t* Wqt = (bf16_t*)(ws + off); off += nW * 2;
  bf16_t* Wkt = (bf16_t*)(ws + off); off += nW * 2;
  bf16_t* Wvt = (bf16_t*)(ws + off); off += nW * 2;
  bf16_t* Wgt = (bf16_t*)(ws + off); off += nW * 2;
  bf16_t* Wot = (bf16_t*)(ws + off); off += nW * 2;
  bf16_t* Qb  = (bf16_t*)(ws + off); off += nX * 2;
  bf16_t* Kb  = (bf16_t*)(ws + off); off += nX * 2;
  bf16_t* Vtb = (bf16_t*)(ws + off); off += nX * 2;
  float*  Gsw = (float*) (ws + off); off += nX * 4;
  bf16_t* Ygb = (bf16_t*)(ws + off); off += nX * 2;
  (void)ws_size; (void)in_sizes; (void)n_in; (void)out_size;

  // 1) precision conversion + weight transposes
  cvt_f32_bf16<<<2048, 256, 0, stream>>>(X, Xb, (int)nX);
  dim3 tb(16, 16), tg(Dc / 16, Dc / 16);
  transpose_cvt<<<tg, tb, 0, stream>>>(Wq, Wqt, Dc);
  transpose_cvt<<<tg, tb, 0, stream>>>(Wk, Wkt, Dc);
  transpose_cvt<<<tg, tb, 0, stream>>>(Wv, Wvt, Dc);
  transpose_cvt<<<tg, tb, 0, stream>>>(Wg, Wgt, Dc);
  transpose_cvt<<<tg, tb, 0, stream>>>(Wo, Wot, Dc);

  // 2) projections: block tile 64(M) x 128(N) -> grid (N/128, M/64)
  dim3 gg(Dc / 128, M / 64);
  const float qscale = 0.08838834764831845f;   // 1/sqrt(DH)
  gemm_bf16_k<<<gg, 256, 0, stream>>>(Xb, Wqt, nullptr, Qb,  M, Dc, Dc, qscale, 1);
  gemm_bf16_k<<<gg, 256, 0, stream>>>(Xb, Wkt, nullptr, Kb,  M, Dc, Dc, 1.0f,   1);
  gemm_bf16_k<<<gg, 256, 0, stream>>>(Xb, Wvt, nullptr, Vtb, M, Dc, Dc, 1.0f,   2);
  gemm_bf16_k<<<gg, 256, 0, stream>>>(Xb, Wgt, Gsw, nullptr, M, Dc, Dc, 1.0f,   3);

  // 3) retention + groupnorm + silu-gate  (B*H*(S/16) = 2048 waves)
  const int rwaves = Bc * Hc * (Sc / 16);
  retention_k<<<rwaves / 4, 128, 0, stream>>>(Qb, Kb, Vtb, Gsw, Ygb);

  // 4) output projection -> f32 d_out
  gemm_bf16_k<<<gg, 256, 0, stream>>>(Ygb, Wot, Out, nullptr, M, Dc, Dc, 1.0f, 0);
}
